// ModelNVDP_49091476193999
// MI455X (gfx1250) — compile-verified, matrix operand order checked by reading, not run on hardware
//
#include <hip/hip_runtime.h>
#include <hip/hip_bf16.h>
#include <stdint.h>

typedef __attribute__((ext_vector_type(16))) __bf16 v16bf;
typedef __attribute__((ext_vector_type(8)))  float  v8f;

constexpr int kB  = 256;
constexpr int kNC = 256;
constexpr int kNT = 256;
constexpr int kXD = 8;
constexpr int kYD = 4;
constexpr int kRD = 128;
constexpr long kM = (long)kB * kNT; // 65536 rows for big GEMMs

// ---------------- device helpers ----------------
__device__ __forceinline__ float sigmoidf_(float x) { return 1.f / (1.f + __expf(-x)); }
__device__ __forceinline__ float clipf_(float x, float lo, float hi) { return fminf(fmaxf(x, lo), hi); }
__device__ __forceinline__ uint32_t mix_(uint32_t x) {
  x ^= x >> 16; x *= 0x7feb352dU; x ^= x >> 15; x *= 0x846ca68bU; x ^= x >> 16; return x;
}
__device__ __forceinline__ float gauss_(uint32_t seed, uint32_t m, uint32_t n) {
  uint32_t h = mix_(seed ^ (0x9E3779B9U * m) ^ (0x85EBCA6BU * n));
  uint32_t g = mix_(h + 0x68E31DA4U);
  float u1 = ((h >> 8) + 1u) * (1.0f / 16777216.0f);
  float u2 = (g >> 8) * (1.0f / 16777216.0f);
  return sqrtf(-2.0f * __logf(u1)) * __cosf(6.28318530718f * u2);
}
__device__ __forceinline__ v8f wmma_bf16(v16bf a, v16bf b, v8f c) {
  return __builtin_amdgcn_wmma_f32_16x16x32_bf16(false, a, false, b, (short)0, c, false, false);
}

// ---- CDNA5 async global->LDS staging (ASYNCcnt-tracked), via inline asm ----
__device__ __forceinline__ uint32_t lds_addr(const void* p) {
  return (uint32_t)(uintptr_t)(__attribute__((address_space(3))) const void*)p;
}
__device__ __forceinline__ void async_b128(uint32_t lds, const void* gaddr) {
  // GLOBAL_LOAD_ASYNC_TO_LDS_B128: vdst = LDS byte address, vaddr = 64-bit global addr
  asm volatile("global_load_async_to_lds_b128 %0, %1, off" :: "v"(lds), "v"(gaddr) : "memory");
}
__device__ __forceinline__ void wait_async3() { asm volatile("s_wait_asynccnt 0x3" ::: "memory"); }
__device__ __forceinline__ void wait_async0() { asm volatile("s_wait_asynccnt 0x0" ::: "memory"); }

// ---------------- generic WMMA GEMM: C = act(A * W^T + bias) ----------------
// A: (rows, Kp) bf16, rows % 128 == 0, Kp % 32 == 0
// W: (Npad64, Kp) bf16 (zero padded rows)
// ACT: 0=none 1=relu 2=leaky(0.1) ; OBF: 1 -> bf16 out, 0 -> f32 out
template <int ACT, int OBF>
__global__ __launch_bounds__(256) void gemm_wmma_k(
    const __bf16* __restrict__ A, const __bf16* __restrict__ W,
    const float* __restrict__ bias, void* __restrict__ Cv,
    int N, int Kp, int ldc) {
  __shared__ alignas(32) __bf16 As[2][128][32];
  __shared__ alignas(32) __bf16 Ws[2][64][32];
  const int tid = threadIdx.x;
  const int wv = tid >> 5, lane = tid & 31, lrow = lane & 15, kh = (lane >> 4) * 16;
  const long rowBase = (long)blockIdx.y * 128;
  const int colBase = blockIdx.x * 64;

  // b128 chunk mapping (8 bf16 each): A = 512 chunks (2/thread), W = 256 chunks (1/thread)
  const int ar0 = tid >> 2;                  // rows 0..63
  const int ar1 = 64 + (tid >> 2);           // rows 64..127
  const int cc  = (tid & 3) * 8;             // col within 32

  auto stage = [&](int buf, int k0) {
    async_b128(lds_addr(&As[buf][ar0][cc]), A + (rowBase + ar0) * (long)Kp + k0 + cc);
    async_b128(lds_addr(&As[buf][ar1][cc]), A + (rowBase + ar1) * (long)Kp + k0 + cc);
    async_b128(lds_addr(&Ws[buf][tid >> 2][cc]), W + (long)(colBase + (tid >> 2)) * Kp + k0 + cc);
  };

  v8f acc[4];
#pragma unroll
  for (int j = 0; j < 4; j++)
#pragma unroll
    for (int e = 0; e < 8; e++) acc[j][e] = 0.f;

  const int S = Kp >> 5;
  stage(0, 0);
  for (int s = 0; s < S; s++) {
    if (s + 1 < S) { stage((s + 1) & 1, (s + 1) << 5); wait_async3(); }
    else           { wait_async0(); }
    __syncthreads();
    const int bu = s & 1;
    v16bf a  = *(const v16bf*)&As[bu][wv * 16 + lrow][kh];
    v16bf b0 = *(const v16bf*)&Ws[bu][ 0 + lrow][kh];
    v16bf b1 = *(const v16bf*)&Ws[bu][16 + lrow][kh];
    v16bf b2 = *(const v16bf*)&Ws[bu][32 + lrow][kh];
    v16bf b3 = *(const v16bf*)&Ws[bu][48 + lrow][kh];
    acc[0] = wmma_bf16(a, b0, acc[0]);
    acc[1] = wmma_bf16(a, b1, acc[1]);
    acc[2] = wmma_bf16(a, b2, acc[2]);
    acc[3] = wmma_bf16(a, b3, acc[3]);
    __syncthreads();
  }
  const int mh = (lane >> 4) * 8;
#pragma unroll
  for (int j = 0; j < 4; j++) {
    int n = colBase + j * 16 + lrow;
    if (n >= N) continue;
    float bi = bias[n];
#pragma unroll
    for (int v = 0; v < 8; v++) {
      long m = rowBase + wv * 16 + mh + v;
      float o = acc[j][v] + bi;
      if (ACT == 1) o = fmaxf(o, 0.f);
      if (ACT == 2) o = (o > 0.f) ? o : 0.1f * o;
      if (OBF) ((__bf16*)Cv)[m * (long)ldc + n] = (__bf16)o;
      else     ((float*)Cv)[m * (long)ldc + n] = o;
    }
  }
}

// ---------------- fused VDP layer (4 simultaneous GEMMs + noise epilogue) ----------------
// mean = acc0 - c*acc1 + meanb ; var = c*acc2 - c^2*acc3 + varb + eps
template <int FINAL>
__global__ __launch_bounds__(256) void vdp_wmma_k(
    const __bf16* __restrict__ X, const __bf16* __restrict__ W,
    const __bf16* __restrict__ W2, const float* __restrict__ s1,
    const float* __restrict__ c1, const float* __restrict__ meanb,
    const float* __restrict__ varb, void* __restrict__ Ov,
    float* __restrict__ mu, float* __restrict__ stdv,
    int Nf, int Kp, int ldo, uint32_t seed) {
  __shared__ alignas(32) __bf16 Xs[2][128][32];
  __shared__ alignas(32) __bf16 Wsh[2][32][32];
  __shared__ alignas(32) __bf16 W2s[2][32][32];
  const int tid = threadIdx.x;
  const int wv = tid >> 5, lane = tid & 31, lrow = lane & 15, kh = (lane >> 4) * 16;
  const long rowBase = (long)blockIdx.y * 128;
  const int colBase = blockIdx.x * 32;
  const int bb = (int)(rowBase / kNT);  // 128 | NT -> whole block in one batch
  const float* s1row = s1 + (long)bb * Kp;

  const int xr0 = tid >> 2;
  const int xr1 = 64 + (tid >> 2);
  const int cc = (tid & 3) * 8;

  // X = 512 chunks (2/thread); W = 128 chunks (waves 0-3); W2 = 128 chunks (waves 4-7)
  auto stage = [&](int buf, int k0) {
    async_b128(lds_addr(&Xs[buf][xr0][cc]), X + (rowBase + xr0) * (long)Kp + k0 + cc);
    async_b128(lds_addr(&Xs[buf][xr1][cc]), X + (rowBase + xr1) * (long)Kp + k0 + cc);
    if (tid < 128) {
      int wr = tid >> 2;
      async_b128(lds_addr(&Wsh[buf][wr][cc]), W + (long)(colBase + wr) * Kp + k0 + cc);
    } else {
      int wr = (tid - 128) >> 2;
      async_b128(lds_addr(&W2s[buf][wr][cc]), W2 + (long)(colBase + wr) * Kp + k0 + cc);
    }
  };

  v8f acc0[2], acc1[2], acc2[2], acc3[2];
#pragma unroll
  for (int j = 0; j < 2; j++)
#pragma unroll
    for (int e = 0; e < 8; e++) { acc0[j][e] = 0.f; acc1[j][e] = 0.f; acc2[j][e] = 0.f; acc3[j][e] = 0.f; }

  const int S = Kp >> 5;
  stage(0, 0);
  for (int s = 0; s < S; s++) {
    const int k0 = s << 5;
    // per-lane gate values for this k-stage (L2-resident, tiny)
    float sv[16];
    *(float4*)&sv[0]  = *(const float4*)(s1row + k0 + kh);
    *(float4*)&sv[4]  = *(const float4*)(s1row + k0 + kh + 4);
    *(float4*)&sv[8]  = *(const float4*)(s1row + k0 + kh + 8);
    *(float4*)&sv[12] = *(const float4*)(s1row + k0 + kh + 12);
    if (s + 1 < S) { stage((s + 1) & 1, (s + 1) << 5); wait_async3(); }
    else           { wait_async0(); }
    __syncthreads();
    const int bu = s & 1;
    v16bf x  = *(const v16bf*)&Xs[bu][wv * 16 + lrow][kh];
    v16bf c0 = *(const v16bf*)&Wsh[bu][ 0 + lrow][kh];
    v16bf c1f = *(const v16bf*)&Wsh[bu][16 + lrow][kh];
    v16bf d0 = *(const v16bf*)&W2s[bu][ 0 + lrow][kh];
    v16bf d1 = *(const v16bf*)&W2s[bu][16 + lrow][kh];
    v16bf a1, a2, a3;
#pragma unroll
    for (int e = 0; e < 16; e++) {
      float xf = (float)x[e];
      float sg = sv[e];
      float xs = xf * sg;
      a1[e] = (__bf16)xs;
      float x2s = xf * xs;      // x^2 * s1
      a2[e] = (__bf16)x2s;
      a3[e] = (__bf16)(x2s * sg);  // x^2 * s1^2
    }
    acc0[0] = wmma_bf16(x,  c0, acc0[0]);
    acc1[0] = wmma_bf16(a1, c0, acc1[0]);
    acc2[0] = wmma_bf16(a2, d0, acc2[0]);
    acc3[0] = wmma_bf16(a3, d0, acc3[0]);
    acc0[1] = wmma_bf16(x,  c1f, acc0[1]);
    acc1[1] = wmma_bf16(a1, c1f, acc1[1]);
    acc2[1] = wmma_bf16(a2, d1, acc2[1]);
    acc3[1] = wmma_bf16(a3, d1, acc3[1]);
    __syncthreads();
  }
  const int mh = (lane >> 4) * 8;
#pragma unroll
  for (int j = 0; j < 2; j++) {
    int n = colBase + j * 16 + lrow;
    if (n >= Nf) continue;
    float c = c1[(long)bb * Nf + n];
    float mb = meanb[(long)bb * Nf + n];
    float vb = varb[(long)bb * Nf + n];
#pragma unroll
    for (int v = 0; v < 8; v++) {
      long m = rowBase + wv * 16 + mh + v;
      float mean = acc0[j][v] - c * acc1[j][v] + mb;
      float var = fmaxf(c * acc2[j][v] - c * c * acc3[j][v] + vb, 0.f) + 1e-10f;
      float z = gauss_(seed, (uint32_t)m, (uint32_t)n);
      float o = mean + sqrtf(var) * z;
      if (!FINAL) {
        o = fmaxf(o, 0.f);
        ((__bf16*)Ov)[m * (long)ldo + n] = (__bf16)o;
      } else {
        if (n < kYD) {
          mu[m * kYD + n] = o;
        } else {
          float sp = (o > 20.f) ? o : log1pf(__expf(o));
          stdv[m * kYD + (n - kYD)] = 0.1f + 0.9f * sp;
        }
      }
    }
  }
}

// ---------------- small utility kernels ----------------
__global__ void pack_enc_k(const float* __restrict__ xc, const float* __restrict__ yc,
                           __bf16* __restrict__ dst) {
  long m = (long)blockIdx.x * blockDim.x + threadIdx.x;  // 65536 rows
  __bf16* d = dst + m * 32;
#pragma unroll
  for (int j = 0; j < 8; j++) d[j] = (__bf16)xc[m * kXD + j];
#pragma unroll
  for (int j = 0; j < 4; j++) d[8 + j] = (__bf16)yc[m * kYD + j];
#pragma unroll
  for (int j = 12; j < 32; j++) d[j] = (__bf16)0.f;
}

__global__ void reduce_mean_k(const float* __restrict__ Ri, float* __restrict__ r32,
                              __bf16* __restrict__ rbf) {
  int b = blockIdx.x, k = threadIdx.x;  // grid 256, block 128
  float s = 0.f;
  for (int i = 0; i < kNC; i++) s += Ri[((long)b * kNC + i) * kRD + k];
  float m = s * (1.0f / kNC);
  r32[b * kRD + k] = m;
  rbf[b * kRD + k] = (__bf16)m;
}

__global__ void pack_concat_k(const float* __restrict__ xt, const float* __restrict__ r32,
                              __bf16* __restrict__ dst) {
  long m = (long)blockIdx.x * blockDim.x + threadIdx.x;
  int b = (int)(m >> 8);
  __bf16* d = dst + m * 160;
#pragma unroll
  for (int j = 0; j < 8; j++) d[j] = (__bf16)xt[m * kXD + j];
  for (int j = 0; j < kRD; j++) d[8 + j] = (__bf16)r32[b * kRD + j];
#pragma unroll
  for (int j = 136; j < 160; j++) d[j] = (__bf16)0.f;
}

__global__ void cvt_w_k(const float* __restrict__ src, __bf16* __restrict__ dst,
                        int rows, int cols, int rowsPad, int colsPad, int sq) {
  size_t i = (size_t)blockIdx.x * blockDim.x + threadIdx.x;
  size_t tot = (size_t)rowsPad * colsPad;
  if (i >= tot) return;
  int r = (int)(i / colsPad), c = (int)(i % colsPad);
  float v = 0.f;
  if (r < rows && c < cols) { v = src[(size_t)r * cols + c]; if (sq) v *= v; }
  dst[i] = (__bf16)v;
}

__global__ void gate_prep_k(const float* __restrict__ la, int ldla,
                            const float* __restrict__ tau, const float* __restrict__ bias,
                            int fin, int fout, int Kp, float* __restrict__ s1,
                            float* __restrict__ c1, float* __restrict__ meanb,
                            float* __restrict__ varb) {
  int b = blockIdx.x;  // 256
  float t = clipf_(tau[0], 0.5f, 5.f);
  float inv_t = 1.f / t;
  const float* lb_ = la + (size_t)b * ldla;
  float s3 = sigmoidf_(clipf_(lb_[fin + fout], -1.f, 2.5f) * inv_t);
  for (int i = threadIdx.x; i < Kp; i += blockDim.x) {
    float v = 0.f;
    if (i < fin) v = sigmoidf_(clipf_(lb_[i], -1.f, 2.5f) * inv_t);
    s1[(size_t)b * Kp + i] = v;
  }
  for (int o = threadIdx.x; o < fout; o += blockDim.x) {
    float s2 = sigmoidf_(clipf_(lb_[fin + o], -1.f, 2.5f) * inv_t);
    float pb = sigmoidf_(clipf_(lb_[fin + fout + 1 + o], -1.f, 2.5f) * inv_t);
    float bo = bias[o];
    c1[(size_t)b * fout + o] = s2 * s3;
    meanb[(size_t)b * fout + o] = (1.f - pb) * bo;
    varb[(size_t)b * fout + o] = pb * (1.f - pb) * bo * bo;
  }
}

// ---------------- host orchestration ----------------
extern "C" void kernel_launch(void* const* d_in, const int* in_sizes, int n_in,
                              void* d_out, int out_size, void* d_ws, size_t ws_size,
                              hipStream_t stream) {
  (void)in_sizes; (void)n_in; (void)out_size; (void)ws_size;
  const float* x_c = (const float*)d_in[0];
  const float* y_c = (const float*)d_in[1];
  const float* x_t = (const float*)d_in[2];
  int ii = 3;
  const float *encW[6], *encB[6];
  for (int l = 0; l < 6; l++) { encW[l] = (const float*)d_in[ii++]; encB[l] = (const float*)d_in[ii++]; }
  struct GP { const float *W, *b, *lW[5], *lb[5], *tau; int fin, fout, Kp, Nla; };
  GP g[4];
  const int fins[4] = {136, 512, 512, 512};
  const int fouts[4] = {512, 512, 512, 8};
  const int kps[4] = {160, 512, 512, 512};
  for (int q = 0; q < 4; q++) {
    g[q].W = (const float*)d_in[ii++];
    g[q].b = (const float*)d_in[ii++];
    for (int l = 0; l < 5; l++) { g[q].lW[l] = (const float*)d_in[ii++]; g[q].lb[l] = (const float*)d_in[ii++]; }
    g[q].tau = (const float*)d_in[ii++];
    g[q].fin = fins[q]; g[q].fout = fouts[q]; g[q].Kp = kps[q];
    g[q].Nla = fins[q] + 2 * fouts[q] + 1;
  }

  char* wsb = (char*)d_ws;
  size_t off = 0;
  auto alloc = [&](size_t bytes) -> void* {
    void* p = wsb + off;
    off += (bytes + 255) & ~(size_t)255;
    return p;
  };
  __bf16* encin = (__bf16*)alloc(kM * 32 * 2);
  __bf16* act0 = (__bf16*)alloc((size_t)kM * 512 * 2);
  __bf16* act1 = (__bf16*)alloc((size_t)kM * 512 * 2);
  float* Ri = (float*)act1;  // alias: act1 is dead while Ri lives
  float* r32 = (float*)alloc(kB * kRD * 4);
  __bf16* rbf = (__bf16*)alloc(kB * kRD * 2);
  __bf16* concat = (__bf16*)alloc((size_t)kM * 160 * 2);
  __bf16* ln0 = (__bf16*)alloc(kB * kRD * 2);
  __bf16* ln1 = (__bf16*)alloc(kB * kRD * 2);
  float* la = (float*)alloc((size_t)kB * 1537 * 4);
  float* s1 = (float*)alloc((size_t)kB * 512 * 4);
  float* c1 = (float*)alloc((size_t)kB * 512 * 4);
  float* meanb = (float*)alloc((size_t)kB * 512 * 4);
  float* varb = (float*)alloc((size_t)kB * 512 * 4);

  auto cvt = [&](const float* src, __bf16* dst, int rows, int cols, int rowsPad, int colsPad, int sq) {
    size_t tot = (size_t)rowsPad * colsPad;
    cvt_w_k<<<dim3((unsigned)((tot + 255) / 256)), dim3(256), 0, stream>>>(src, dst, rows, cols, rowsPad, colsPad, sq);
  };

  __bf16* eW[6];
  eW[0] = (__bf16*)alloc(512 * 32 * 2); cvt(encW[0], eW[0], 512, 12, 512, 32, 0);
  for (int l = 1; l < 5; l++) { eW[l] = (__bf16*)alloc((size_t)512 * 512 * 2); cvt(encW[l], eW[l], 512, 512, 512, 512, 0); }
  eW[5] = (__bf16*)alloc((size_t)128 * 512 * 2); cvt(encW[5], eW[5], 128, 512, 128, 512, 0);

  __bf16 *lwbf[4][5], *mwbf[4], *mw2bf[4];
  int nla64[4], nfpad[4];
  for (int q = 0; q < 4; q++) {
    for (int l = 0; l < 4; l++) {
      lwbf[q][l] = (__bf16*)alloc((size_t)128 * 128 * 2);
      cvt(g[q].lW[l], lwbf[q][l], 128, 128, 128, 128, 0);
    }
    nla64[q] = ((g[q].Nla + 63) / 64) * 64;
    lwbf[q][4] = (__bf16*)alloc((size_t)nla64[q] * 128 * 2);
    cvt(g[q].lW[4], lwbf[q][4], g[q].Nla, 128, nla64[q], 128, 0);
    nfpad[q] = ((g[q].fout + 31) / 32) * 32;
    mwbf[q] = (__bf16*)alloc((size_t)nfpad[q] * g[q].Kp * 2);
    mw2bf[q] = (__bf16*)alloc((size_t)nfpad[q] * g[q].Kp * 2);
    cvt(g[q].W, mwbf[q], g[q].fout, g[q].fin, nfpad[q], g[q].Kp, 0);
    cvt(g[q].W, mw2bf[q], g[q].fout, g[q].fin, nfpad[q], g[q].Kp, 1);
  }

  auto gemm = [&](const __bf16* A, const __bf16* W, const float* bias, void* C,
                  long Mrows, int N, int Kp, int ldc, int act, int obf) {
    dim3 gr((unsigned)((N + 63) / 64), (unsigned)(Mrows / 128)), bl(256);
    if (act == 1 && obf == 1)      gemm_wmma_k<1, 1><<<gr, bl, 0, stream>>>(A, W, bias, C, N, Kp, ldc);
    else if (act == 2 && obf == 1) gemm_wmma_k<2, 1><<<gr, bl, 0, stream>>>(A, W, bias, C, N, Kp, ldc);
    else                           gemm_wmma_k<0, 0><<<gr, bl, 0, stream>>>(A, W, bias, C, N, Kp, ldc);
  };

  // ---- encoder ----
  pack_enc_k<<<dim3(256), dim3(256), 0, stream>>>(x_c, y_c, encin);
  gemm(encin, eW[0], encB[0], act0, kM, 512, 32, 512, 1, 1);
  gemm(act0, eW[1], encB[1], act1, kM, 512, 512, 512, 1, 1);
  gemm(act1, eW[2], encB[2], act0, kM, 512, 512, 512, 1, 1);
  gemm(act0, eW[3], encB[3], act1, kM, 512, 512, 512, 1, 1);
  gemm(act1, eW[4], encB[4], act0, kM, 512, 512, 512, 1, 1);
  gemm(act0, eW[5], encB[5], Ri, kM, 128, 512, 128, 0, 0);
  reduce_mean_k<<<dim3(256), dim3(128), 0, stream>>>(Ri, r32, rbf);
  pack_concat_k<<<dim3(256), dim3(256), 0, stream>>>(x_t, r32, concat);

  // ---- 4 VDP layers ----
  float* mu = (float*)d_out;
  float* sd = (float*)d_out + (size_t)kM * kYD;
  const __bf16* xin[4] = {concat, act0, act1, act0};
  __bf16* xout[4] = {act0, act1, act0, nullptr};
  for (int q = 0; q < 4; q++) {
    gemm(rbf, lwbf[q][0], g[q].lb[0], ln0, kB, 128, 128, 128, 2, 1);
    gemm(ln0, lwbf[q][1], g[q].lb[1], ln1, kB, 128, 128, 128, 2, 1);
    gemm(ln1, lwbf[q][2], g[q].lb[2], ln0, kB, 128, 128, 128, 2, 1);
    gemm(ln0, lwbf[q][3], g[q].lb[3], ln1, kB, 128, 128, 128, 2, 1);
    gemm(ln1, lwbf[q][4], g[q].lb[4], la, kB, g[q].Nla, 128, g[q].Nla, 0, 0);
    gate_prep_k<<<dim3(256), dim3(256), 0, stream>>>(la, g[q].Nla, g[q].tau, g[q].b,
                                                     g[q].fin, g[q].fout, g[q].Kp,
                                                     s1, c1, meanb, varb);
    dim3 gr((unsigned)((g[q].fout + 31) / 32), (unsigned)(kM / 128)), bl(256);
    uint32_t seed = 0xC0FFEEu + 0x9E37u * (uint32_t)q;
    if (q < 3)
      vdp_wmma_k<0><<<gr, bl, 0, stream>>>(xin[q], mwbf[q], mw2bf[q], s1, c1, meanb, varb,
                                           xout[q], nullptr, nullptr, g[q].fout, g[q].Kp, 512, seed);
    else
      vdp_wmma_k<1><<<gr, bl, 0, stream>>>(xin[q], mwbf[q], mw2bf[q], s1, c1, meanb, varb,
                                           nullptr, mu, sd, g[q].fout, g[q].Kp, 0, seed);
  }
}
